// LorentzGNN_17781164605815
// MI455X (gfx1250) — compile-verified, matrix operand order checked by reading, not run on hardware
//
#include <hip/hip_runtime.h>
#include <hip/hip_bf16.h>
#include <cstdint>
#include <cstddef>

typedef __attribute__((ext_vector_type(16))) __bf16 bf16x16;
typedef __attribute__((ext_vector_type(8)))  __bf16 bf16x8;
typedef __attribute__((ext_vector_type(8)))  float  f32x8;

#define NNODES 50000
#define NEDGES 600000
#define FT     128
#define HEADS  4
#define CCH    32
#define NEG_SLOPE 0.2f
#define BATCH  100
#define DIN    129   // FT_OUT+1
#define DMID   513   // 4*HID+1

// ---------------- utility kernels ----------------

__global__ void zero_f32(float* __restrict__ p, int n) {
    int i = blockIdx.x * 256 + threadIdx.x;
    if (i < n) p[i] = 0.0f;
}

__global__ void fill_i32(int* __restrict__ p, int v, int n) {
    int i = blockIdx.x * 256 + threadIdx.x;
    if (i < n) p[i] = v;
}

__global__ void fix_src(const int* __restrict__ src0, const int* __restrict__ dst0,
                        int* __restrict__ srcfix, int ne, int nn) {
    int i = blockIdx.x * 256 + threadIdx.x;
    if (i < ne) {
        int s = src0[i], d = dst0[i];
        srcfix[i] = (s == d) ? ((s + 1) % nn) : s;
    }
}

__global__ void x_to_bf16(const float* __restrict__ x, __bf16* __restrict__ xbf, int nn) {
    int i = blockIdx.x * 256 + threadIdx.x;
    if (i < nn * FT) {
        int node = i >> 7, c = i & 127;
        xbf[i] = (__bf16)x[(size_t)node * (FT + 1) + 1 + c];
    }
}

__global__ void w_to_bf16(const float* __restrict__ w, __bf16* __restrict__ wbf, int n) {
    int i = blockIdx.x * 256 + threadIdx.x;
    if (i < n) wbf[i] = (__bf16)w[i];
}

// ---------------- WMMA GEMM: Y[n,128] = Xbf[n,128] @ Wbf[128,128] + bias ----------------
// block = 256 threads = 8 waves; each wave computes 16 rows x 128 cols.
// A frag (16x32 bf16) lane mapping (ISA 7.12.2): lane l<16 -> row l, K chunks {0..7,16..23};
// lane l>=16 -> row l-16, K chunks {8..15,24..31}. B frag symmetric with lane = column.
__global__ __launch_bounds__(256) void gemm_bias_wmma(
        const __bf16* __restrict__ X, const __bf16* __restrict__ W,
        const float* __restrict__ bias, float* __restrict__ Y, int nrows)
{
    __shared__ __attribute__((aligned(16))) __bf16 Wt[FT * FT]; // transposed: Wt[n*128+k]
    int t = threadIdx.x;
    for (int e = t; e < FT * FT; e += 256) {
        int k = e >> 7, n = e & 127;
        Wt[n * FT + k] = W[e];
    }
    __syncthreads();

    int wave = t >> 5, lane = t & 31;
    int row0 = blockIdx.x * 128 + wave * 16;
    int r = row0 + (lane & 15);
    int rc = (r < nrows) ? r : (nrows - 1);
    int kchunk = (lane >= 16) ? 8 : 0;

    bf16x16 afrag[4];
#pragma unroll
    for (int ks = 0; ks < 4; ks++) {
        const __bf16* xp = X + (size_t)rc * FT + ks * 32 + kchunk;
        bf16x8 lo = *(const bf16x8*)(xp);
        bf16x8 hi = *(const bf16x8*)(xp + 16);
#pragma unroll
        for (int i = 0; i < 8; i++) { afrag[ks][i] = lo[i]; afrag[ks][8 + i] = hi[i]; }
    }

    int col_l = lane & 15;
    int rbase = row0 + ((lane >= 16) ? 8 : 0);
#pragma unroll
    for (int nt = 0; nt < 8; nt++) {
        f32x8 acc = {0.f, 0.f, 0.f, 0.f, 0.f, 0.f, 0.f, 0.f};
#pragma unroll
        for (int ks = 0; ks < 4; ks++) {
            const __bf16* wp = &Wt[(size_t)(nt * 16 + col_l) * FT + ks * 32 + kchunk];
            bf16x8 lo = *(const bf16x8*)(wp);
            bf16x8 hi = *(const bf16x8*)(wp + 16);
            bf16x16 bfrag;
#pragma unroll
            for (int i = 0; i < 8; i++) { bfrag[i] = lo[i]; bfrag[8 + i] = hi[i]; }
            acc = __builtin_amdgcn_wmma_f32_16x16x32_bf16(
                false, afrag[ks], false, bfrag, (short)0, acc, false, false);
        }
        int col = nt * 16 + col_l;
        float bv = bias[col];
#pragma unroll
        for (int v = 0; v < 8; v++) {
            int rr = rbase + v;
            if (rr < nrows) Y[(size_t)rr * FT + col] = acc[v] + bv;
        }
    }
}

// ---------------- GAT edge kernels ----------------

__device__ __forceinline__ int fkey(float f) {
    int b = __float_as_int(f);
    return (b >= 0) ? b : (b ^ 0x7fffffff);
}
__device__ __forceinline__ float finv(int k) {
    int b = (k >= 0) ? k : (k ^ 0x7fffffff);
    return __int_as_float(b);
}

// wave per edge: alpha[e][h] = sum_c att[h*32+c] * leaky_relu(xl[src][..] + xr[dst][..])
__global__ __launch_bounds__(256) void edge_alpha(
        const int* __restrict__ srcv, const int* __restrict__ dstv,
        const float* __restrict__ xl, const float* __restrict__ xr,
        const float* __restrict__ att, float* __restrict__ alpha, int ne)
{
    int wave = threadIdx.x >> 5, lane = threadIdx.x & 31;
    int e = blockIdx.x * 8 + wave;
    if (e >= ne) return;
    int s = srcv[e], d = dstv[e];
    int c0 = lane * 4;
    float4 a = *(const float4*)(xl + (size_t)s * FT + c0);
    float4 b = *(const float4*)(xr + (size_t)d * FT + c0);
    float4 w = *(const float4*)(att + c0);
    float vs[4] = {a.x + b.x, a.y + b.y, a.z + b.z, a.w + b.w};
    float ww[4] = {w.x, w.y, w.z, w.w};
    float sum = 0.f;
#pragma unroll
    for (int j = 0; j < 4; j++) {
        float v = vs[j];
        v = (v > 0.f) ? v : v * NEG_SLOPE;
        sum += ww[j] * v;
    }
    sum += __shfl_xor(sum, 1);
    sum += __shfl_xor(sum, 2);
    sum += __shfl_xor(sum, 4);
    if ((lane & 7) == 0) alpha[(size_t)e * HEADS + (lane >> 3)] = sum;
}

// wave per node: self-loop alpha
__global__ __launch_bounds__(256) void self_alpha(
        const float* __restrict__ xl, const float* __restrict__ xr,
        const float* __restrict__ att, float* __restrict__ aself, int nn)
{
    int wave = threadIdx.x >> 5, lane = threadIdx.x & 31;
    int i = blockIdx.x * 8 + wave;
    if (i >= nn) return;
    int c0 = lane * 4;
    float4 a = *(const float4*)(xl + (size_t)i * FT + c0);
    float4 b = *(const float4*)(xr + (size_t)i * FT + c0);
    float4 w = *(const float4*)(att + c0);
    float vs[4] = {a.x + b.x, a.y + b.y, a.z + b.z, a.w + b.w};
    float ww[4] = {w.x, w.y, w.z, w.w};
    float sum = 0.f;
#pragma unroll
    for (int j = 0; j < 4; j++) {
        float v = vs[j];
        v = (v > 0.f) ? v : v * NEG_SLOPE;
        sum += ww[j] * v;
    }
    sum += __shfl_xor(sum, 1);
    sum += __shfl_xor(sum, 2);
    sum += __shfl_xor(sum, 4);
    if ((lane & 7) == 0) aself[(size_t)i * HEADS + (lane >> 3)] = sum;
}

__global__ void edge_max(const int* __restrict__ dstv, const float* __restrict__ alpha,
                         int* __restrict__ amaxk, int ne) {
    int idx = blockIdx.x * 256 + threadIdx.x;
    if (idx < ne * HEADS) {
        int e = idx >> 2, h = idx & 3;
        int d = dstv[e];
        atomicMax(&amaxk[d * HEADS + h], fkey(alpha[idx]));
    }
}

__global__ void self_max(const float* __restrict__ aself, int* __restrict__ amaxk, int nn) {
    int idx = blockIdx.x * 256 + threadIdx.x;
    if (idx < nn * HEADS) {
        int k = fkey(aself[idx]);
        if (k > amaxk[idx]) amaxk[idx] = k; // ordered after edge_max; only this thread touches idx
    }
}

__global__ void edge_expsum(const int* __restrict__ dstv, float* __restrict__ alpha,
                            const int* __restrict__ amaxk, float* __restrict__ den, int ne) {
    int idx = blockIdx.x * 256 + threadIdx.x;
    if (idx < ne * HEADS) {
        int e = idx >> 2, h = idx & 3;
        int d = dstv[e];
        float m = finv(amaxk[d * HEADS + h]);
        float ex = __expf(alpha[idx] - m);
        alpha[idx] = ex;
        atomicAdd(&den[d * HEADS + h], ex);
    }
}

__global__ void self_expsum(float* __restrict__ aself, const int* __restrict__ amaxk,
                            float* __restrict__ den, int nn) {
    int idx = blockIdx.x * 256 + threadIdx.x;
    if (idx < nn * HEADS) {
        float ex = __expf(aself[idx] - finv(amaxk[idx]));
        aself[idx] = ex;
        den[idx] += ex; // ordered after edge_expsum
    }
}

// wave per edge: out[dst] += xl[src] * a
__global__ __launch_bounds__(256) void edge_aggregate(
        const int* __restrict__ srcv, const int* __restrict__ dstv,
        const float* __restrict__ xl, const float* __restrict__ alpha,
        const float* __restrict__ den, float* __restrict__ out, int ne)
{
    int wave = threadIdx.x >> 5, lane = threadIdx.x & 31;
    int e = blockIdx.x * 8 + wave;
    if (e >= ne) return;
    int s = srcv[e], d = dstv[e];
    int c0 = lane * 4;
    int h = lane >> 3;
    float a = alpha[(size_t)e * HEADS + h] / (den[d * HEADS + h] + 1e-16f);
    float4 xv = *(const float4*)(xl + (size_t)s * FT + c0);
    float* op = out + (size_t)d * FT + c0;
    atomicAdd(op + 0, xv.x * a);
    atomicAdd(op + 1, xv.y * a);
    atomicAdd(op + 2, xv.z * a);
    atomicAdd(op + 3, xv.w * a);
}

__global__ __launch_bounds__(256) void self_aggregate(
        const float* __restrict__ xl, const float* __restrict__ aself,
        const float* __restrict__ den, float* __restrict__ out, int nn)
{
    int wave = threadIdx.x >> 5, lane = threadIdx.x & 31;
    int i = blockIdx.x * 8 + wave;
    if (i >= nn) return;
    int c0 = lane * 4;
    int h = lane >> 3;
    float a = aself[(size_t)i * HEADS + h] / (den[i * HEADS + h] + 1e-16f);
    float4 xv = *(const float4*)(xl + (size_t)i * FT + c0);
    float* op = out + (size_t)i * FT + c0;
    op[0] += xv.x * a; op[1] += xv.y * a; op[2] += xv.z * a; op[3] += xv.w * a;
}

// gelu(out + bias) -> bf16 activations for next layer
__global__ void gelu_to_bf16(const float* __restrict__ agg, const float* __restrict__ bias,
                             __bf16* __restrict__ xbf, int n) {
    int i = blockIdx.x * 256 + threadIdx.x;
    if (i < n) {
        float v = agg[i] + bias[i & 127];
        float g = 0.5f * v * (1.f + erff(v * 0.70710678118f));
        xbf[i] = (__bf16)g;
    }
}

// layer-2 finalize: v = agg + bias2; t = sqrt(sum v^2 + 1); accumulate centroid; extract batch rows
__global__ __launch_bounds__(256) void finalize_gat2(
        const float* __restrict__ agg, const float* __restrict__ bias2,
        float* __restrict__ rootacc, float* __restrict__ hsel, int nn)
{
    __shared__ float lacc[DIN];
    int t = threadIdx.x;
    for (int j = t; j < DIN; j += 256) lacc[j] = 0.f;
    __syncthreads();
    int wave = t >> 5, lane = t & 31;
    int i = blockIdx.x * 8 + wave;
    if (i < nn) {
        int c0 = lane * 4;
        float4 av = *(const float4*)(agg + (size_t)i * FT + c0);
        float4 bv = *(const float4*)(bias2 + c0);
        float v[4] = {av.x + bv.x, av.y + bv.y, av.z + bv.z, av.w + bv.w};
        float ss = v[0]*v[0] + v[1]*v[1] + v[2]*v[2] + v[3]*v[3];
        ss += __shfl_xor(ss, 1);
        ss += __shfl_xor(ss, 2);
        ss += __shfl_xor(ss, 4);
        ss += __shfl_xor(ss, 8);
        ss += __shfl_xor(ss, 16);
        float tt = sqrtf(ss + 1.0f); // K = 1
#pragma unroll
        for (int j = 0; j < 4; j++) atomicAdd(&lacc[1 + c0 + j], v[j]);
        if (lane == 0) atomicAdd(&lacc[0], tt);
        if (i % (NNODES / BATCH) == 0) {
            int b = i / (NNODES / BATCH);
#pragma unroll
            for (int j = 0; j < 4; j++) hsel[(size_t)b * DIN + 1 + c0 + j] = v[j];
            if (lane == 0) hsel[(size_t)b * DIN] = tt;
        }
    }
    __syncthreads();
    for (int j = t; j < DIN; j += 256) atomicAdd(&rootacc[j], lacc[j]);
}

__global__ void root_finalize(const float* __restrict__ rootacc, float* __restrict__ outroot, int nn) {
    if (threadIdx.x == 0 && blockIdx.x == 0) {
        float avg[DIN];
        for (int j = 0; j < DIN; j++) avg[j] = rootacc[j] / (float)nn;
        float inner = -avg[0] * avg[0];
        for (int j = 1; j < DIN; j++) inner += avg[j] * avg[j];
        float denom = sqrtf(fmaxf(fabsf(inner), 1e-8f));
        for (int j = 0; j < DIN; j++) outroot[j] = avg[j] / denom; // sqrt(K)=1
    }
}

// ---------------- small Lorentz MLP ----------------

__global__ void small_linear(const float* __restrict__ in, const float* __restrict__ W,
                             const float* __restrict__ b, float* __restrict__ out,
                             int M, int Kd, int Nd) {
    int idx = blockIdx.x * 256 + threadIdx.x;
    if (idx >= M * Nd) return;
    int m = idx / Nd, n = idx % Nd;
    float acc = b[n];
    const float* ip = in + (size_t)m * Kd;
    for (int k = 0; k < Kd; k++) acc += ip[k] * W[(size_t)k * Nd + n];
    out[idx] = acc;
}

__device__ __forceinline__ float block_reduce_sum(float v) {
    __shared__ float red[8];
    int lane = threadIdx.x & 31, w = threadIdx.x >> 5;
    v += __shfl_xor(v, 1);
    v += __shfl_xor(v, 2);
    v += __shfl_xor(v, 4);
    v += __shfl_xor(v, 8);
    v += __shfl_xor(v, 16);
    if (lane == 0) red[w] = v;
    __syncthreads();
    float t = (threadIdx.x < 8) ? red[threadIdx.x] : 0.f;
    if (w == 0) {
        t += __shfl_xor(t, 1);
        t += __shfl_xor(t, 2);
        t += __shfl_xor(t, 4);
        if (lane == 0) red[0] = t;
    }
    __syncthreads();
    return red[0];
}

// in-place lorentz fixup of y[M,D]: time = sigmoid(y0)*exp(s)+1.1; space *= sqrt((time^2-1)/sumsq)
__global__ __launch_bounds__(256) void lorentz_fix(float* __restrict__ y,
                                                   const float* __restrict__ s_log, int D) {
    int row = blockIdx.x, t = threadIdx.x;
    float* yr = y + (size_t)row * D;
    float ss = 0.f;
    for (int j = 1 + t; j < D; j += 256) { float v = yr[j]; ss += v * v; }
    float total = block_reduce_sum(ss);
    float y0 = yr[0];
    float tim = 1.f / (1.f + __expf(-y0)) * __expf(s_log[0]) + 1.1f;
    float sq = fmaxf(total, 1e-8f);
    float scale = sqrtf((tim * tim - 1.0f) / sq);
    __syncthreads();
    for (int j = 1 + t; j < D; j += 256) yr[j] *= scale;
    if (t == 0) yr[0] = tim;
}

// t2[row] = add_time(gelu(t1[row][1:DMID]))
__global__ __launch_bounds__(256) void gelu_addtime(const float* __restrict__ t1,
                                                    float* __restrict__ t2) {
    int row = blockIdx.x, t = threadIdx.x;
    const float* ir = t1 + (size_t)row * DMID;
    float* orr = t2 + (size_t)row * DMID;
    float ss = 0.f;
    for (int j = t; j < DMID - 1; j += 256) {
        float v = ir[1 + j];
        float g = 0.5f * v * (1.f + erff(v * 0.70710678118f));
        orr[1 + j] = g;
        ss += g * g;
    }
    float total = block_reduce_sum(ss);
    if (t == 0) orr[0] = sqrtf(total + 1.0f);
}

// ---------------- host orchestration ----------------

extern "C" void kernel_launch(void* const* d_in, const int* in_sizes, int n_in,
                              void* d_out, int out_size, void* d_ws, size_t ws_size,
                              hipStream_t stream) {
    const float* x    = (const float*)d_in[0];
    const int*   ei   = (const int*)d_in[1];
    const float* W1l  = (const float*)d_in[2];
    const float* b1l  = (const float*)d_in[3];
    const float* W1r  = (const float*)d_in[4];
    const float* b1r  = (const float*)d_in[5];
    const float* att1 = (const float*)d_in[6];
    const float* bias1= (const float*)d_in[7];
    const float* W2l  = (const float*)d_in[8];
    const float* b2l  = (const float*)d_in[9];
    const float* W2r  = (const float*)d_in[10];
    const float* b2r  = (const float*)d_in[11];
    const float* att2 = (const float*)d_in[12];
    const float* bias2= (const float*)d_in[13];
    const float* Wl1  = (const float*)d_in[14];
    const float* bl1  = (const float*)d_in[15];
    const float* s1   = (const float*)d_in[16];
    const float* Wl2  = (const float*)d_in[17];
    const float* bl2  = (const float*)d_in[18];
    const float* s2   = (const float*)d_in[19];
    const float* Wf   = (const float*)d_in[20];
    const float* bfin = (const float*)d_in[21];
    const float* sf   = (const float*)d_in[22];

    const int* src0 = ei;
    const int* dst0 = ei + NEDGES;
    float* out = (float*)d_out;

    // workspace bump allocator
    char* wsp = (char*)d_ws;
    auto alloc = [&](size_t bytes) -> void* {
        void* p = (void*)wsp;
        wsp += (bytes + 255) & ~(size_t)255;
        return p;
    };
    __bf16* xbf    = (__bf16*)alloc((size_t)NNODES * FT * 2);
    __bf16* wlbf   = (__bf16*)alloc((size_t)FT * FT * 2);
    __bf16* wrbf   = (__bf16*)alloc((size_t)FT * FT * 2);
    float*  xl     = (float*)alloc((size_t)NNODES * FT * 4);
    float*  xr     = (float*)alloc((size_t)NNODES * FT * 4);
    int*    srcfix = (int*)alloc((size_t)NEDGES * 4);
    float*  alpha  = (float*)alloc((size_t)NEDGES * HEADS * 4);
    float*  aself  = (float*)alloc((size_t)NNODES * HEADS * 4);
    int*    amaxk  = (int*)alloc((size_t)NNODES * HEADS * 4);
    float*  den    = (float*)alloc((size_t)NNODES * HEADS * 4);
    float*  agg    = (float*)alloc((size_t)NNODES * FT * 4);
    float*  rootacc= (float*)alloc(DIN * 4);
    float*  hsel   = (float*)alloc((size_t)BATCH * DIN * 4);
    float*  t1     = (float*)alloc((size_t)BATCH * DMID * 4);
    float*  t2     = (float*)alloc((size_t)BATCH * DMID * 4);
    float*  t3     = (float*)alloc((size_t)BATCH * DIN * 4);

    dim3 b256(256);
    int gElem  = (NNODES * FT + 255) / 256;
    int gEdgeW = (NEDGES + 7) / 8;       // wave-per-edge
    int gNodeW = (NNODES + 7) / 8;       // wave-per-node
    int gEdgeH = (NEDGES * HEADS + 255) / 256;
    int gNodeH = (NNODES * HEADS + 255) / 256;
    int gGemm  = (NNODES + 127) / 128;

    // ---- preprocessing ----
    fix_src<<<(NEDGES + 255) / 256, b256, 0, stream>>>(src0, dst0, srcfix, NEDGES, NNODES);
    x_to_bf16<<<gElem, b256, 0, stream>>>(x, xbf, NNODES);

    // ================= GAT layer 1 =================
    w_to_bf16<<<(FT * FT + 255) / 256, b256, 0, stream>>>(W1l, wlbf, FT * FT);
    w_to_bf16<<<(FT * FT + 255) / 256, b256, 0, stream>>>(W1r, wrbf, FT * FT);
    gemm_bias_wmma<<<gGemm, b256, 0, stream>>>(xbf, wlbf, b1l, xl, NNODES);
    gemm_bias_wmma<<<gGemm, b256, 0, stream>>>(xbf, wrbf, b1r, xr, NNODES);

    edge_alpha<<<gEdgeW, b256, 0, stream>>>(srcfix, dst0, xl, xr, att1, alpha, NEDGES);
    self_alpha<<<gNodeW, b256, 0, stream>>>(xl, xr, att1, aself, NNODES);
    fill_i32<<<gNodeH, b256, 0, stream>>>(amaxk, (int)0x80000000, NNODES * HEADS);
    edge_max<<<gEdgeH, b256, 0, stream>>>(dst0, alpha, amaxk, NEDGES);
    self_max<<<gNodeH, b256, 0, stream>>>(aself, amaxk, NNODES);
    zero_f32<<<gNodeH, b256, 0, stream>>>(den, NNODES * HEADS);
    edge_expsum<<<gEdgeH, b256, 0, stream>>>(dst0, alpha, amaxk, den, NEDGES);
    self_expsum<<<gNodeH, b256, 0, stream>>>(aself, amaxk, den, NNODES);
    zero_f32<<<gElem, b256, 0, stream>>>(agg, NNODES * FT);
    edge_aggregate<<<gEdgeW, b256, 0, stream>>>(srcfix, dst0, xl, alpha, den, agg, NEDGES);
    self_aggregate<<<gNodeW, b256, 0, stream>>>(xl, aself, den, agg, NNODES);
    gelu_to_bf16<<<gElem, b256, 0, stream>>>(agg, bias1, xbf, NNODES * FT);

    // ================= GAT layer 2 =================
    w_to_bf16<<<(FT * FT + 255) / 256, b256, 0, stream>>>(W2l, wlbf, FT * FT);
    w_to_bf16<<<(FT * FT + 255) / 256, b256, 0, stream>>>(W2r, wrbf, FT * FT);
    gemm_bias_wmma<<<gGemm, b256, 0, stream>>>(xbf, wlbf, b2l, xl, NNODES);
    gemm_bias_wmma<<<gGemm, b256, 0, stream>>>(xbf, wrbf, b2r, xr, NNODES);

    edge_alpha<<<gEdgeW, b256, 0, stream>>>(srcfix, dst0, xl, xr, att2, alpha, NEDGES);
    self_alpha<<<gNodeW, b256, 0, stream>>>(xl, xr, att2, aself, NNODES);
    fill_i32<<<gNodeH, b256, 0, stream>>>(amaxk, (int)0x80000000, NNODES * HEADS);
    edge_max<<<gEdgeH, b256, 0, stream>>>(dst0, alpha, amaxk, NEDGES);
    self_max<<<gNodeH, b256, 0, stream>>>(aself, amaxk, NNODES);
    zero_f32<<<gNodeH, b256, 0, stream>>>(den, NNODES * HEADS);
    edge_expsum<<<gEdgeH, b256, 0, stream>>>(dst0, alpha, amaxk, den, NEDGES);
    self_expsum<<<gNodeH, b256, 0, stream>>>(aself, amaxk, den, NNODES);
    zero_f32<<<gElem, b256, 0, stream>>>(agg, NNODES * FT);
    edge_aggregate<<<gEdgeW, b256, 0, stream>>>(srcfix, dst0, xl, alpha, den, agg, NEDGES);
    self_aggregate<<<gNodeW, b256, 0, stream>>>(xl, aself, den, agg, NNODES);

    // ---- add_time + centroid + batch slice ----
    zero_f32<<<1, b256, 0, stream>>>(rootacc, DIN);
    finalize_gat2<<<gNodeW, b256, 0, stream>>>(agg, bias2, rootacc, hsel, NNODES);
    root_finalize<<<1, b256, 0, stream>>>(rootacc, out + (size_t)BATCH * DIN, NNODES);

    // ---- Lorentz MLP on (100, 129) ----
    small_linear<<<(BATCH * DMID + 255) / 256, b256, 0, stream>>>(hsel, Wl1, bl1, t1, BATCH, DIN, DMID);
    lorentz_fix<<<BATCH, b256, 0, stream>>>(t1, s1, DMID);
    gelu_addtime<<<BATCH, b256, 0, stream>>>(t1, t2);
    small_linear<<<(BATCH * DIN + 255) / 256, b256, 0, stream>>>(t2, Wl2, bl2, t3, BATCH, DMID, DIN);
    lorentz_fix<<<BATCH, b256, 0, stream>>>(t3, s2, DIN);
    small_linear<<<(BATCH * DIN + 255) / 256, b256, 0, stream>>>(t3, Wf, bfin, out, BATCH, DIN, DIN);
    lorentz_fix<<<BATCH, b256, 0, stream>>>(out, sf, DIN);

    (void)in_sizes; (void)n_in; (void)out_size; (void)ws_size;
}
